// RefineModule_85976655331345
// MI455X (gfx1250) — compile-verified, matrix operand order checked by reading, not run on hardware
//
#include <hip/hip_runtime.h>
#include <hip/hip_bf16.h>
#include <math.h>

#define BB 8
#define NN 4096
#define DD 64
#define KK 16
#define CIN 131
#define CPAD 160           // CIN padded to 5 chunks of 32
#define HID 32
#define EPSV 1e-5f
#define SLOPE 0.1f

typedef __attribute__((ext_vector_type(16))) _Float16 v16h;
typedef __attribute__((ext_vector_type(8)))  float    v8f;

// ---------------------------------------------------------------------------
// Kernel 1: KNN.  One thread per query point. Candidates tiled via LDS.
// Blocks tile within a single batch (16 blocks of 256 per batch).
// ---------------------------------------------------------------------------
__global__ void knn_kernel(const float* __restrict__ pc1, int* __restrict__ kidx) {
    __shared__ float tx[512], ty[512], tz[512];
    const int b = blockIdx.x >> 4;
    const int n = ((blockIdx.x & 15) << 8) + threadIdx.x;
    const float* pb = pc1 + (size_t)b * 3 * NN;
    const float px = pb[n], py = pb[NN + n], pz = pb[2 * NN + n];

    float bestd[KK];
    int   besti[KK];
#pragma unroll
    for (int i = 0; i < KK; ++i) { bestd[i] = 3.4e38f; besti[i] = i; }
    float wv = 3.4e38f;
    int   wslot = 0;

    for (int tile = 0; tile < NN; tile += 512) {
        __syncthreads();
        for (int t = threadIdx.x; t < 512; t += 256) {
            tx[t] = pb[tile + t];
            ty[t] = pb[NN + tile + t];
            tz[t] = pb[2 * NN + tile + t];
        }
        __syncthreads();
        if (tile + 512 < NN) __builtin_prefetch(pb + tile + 512, 0, 1);

        for (int j = 0; j < 512; ++j) {
            const float dx = tx[j] - px, dy = ty[j] - py, dz = tz[j] - pz;
            const float d = fmaf(dx, dx, fmaf(dy, dy, dz * dz));
            if (d < wv) {
                const int jj = tile + j;
#pragma unroll
                for (int i = 0; i < KK; ++i)
                    if (i == wslot) { bestd[i] = d; besti[i] = jj; }
                wv = -1.0f;
#pragma unroll
                for (int i = 0; i < KK; ++i)
                    if (bestd[i] > wv) { wv = bestd[i]; wslot = i; }
            }
        }
    }
    int* out = kidx + ((size_t)b * NN + n) * KK;
#pragma unroll
    for (int i = 0; i < KK; ++i) out[i] = besti[i];
}

// ---------------------------------------------------------------------------
// Kernel 2: fusion build (LDS, f16) + MLP layer 1 via WMMA f32_16x16x32_f16.
// One wave per (b,n): A = 16x160 fusion (rows = 16 neighbors),
// B = W1^T in two 16-col tiles, 5 K-chunks -> 10 WMMAs/point, f32 accum.
// Also accumulates per-channel sum / sumsq for BatchNorm.
// ---------------------------------------------------------------------------
__global__ void fusion_wmma_kernel(const float* __restrict__ pc1,
                                   const float* __restrict__ feat1,
                                   const int*   __restrict__ kidx,
                                   const float* __restrict__ W1,
                                   const float* __restrict__ b1,
                                   float* __restrict__ hbuf,
                                   float* __restrict__ stats) {
    __shared__ _Float16 fus[8][KK][CPAD];     // 40960 B
    __shared__ _Float16 w1s[HID][CPAD];       // 10240 B
    __shared__ int      kid[8][KK];
    __shared__ float    ssum[HID], ssq[HID];

    const int tid  = threadIdx.x;
    const int w    = tid >> 5;
    const int lane = tid & 31;
    const int lo   = lane & 15;
    const int hi   = lane >> 4;

    const int pn = blockIdx.x * 8 + w;
    const int b  = pn >> 12;           // / NN
    const int n  = pn & (NN - 1);

    // Stage 1: W1 -> LDS (padded with zeros), neighbor ids, zero stat slots.
    for (int idx = tid; idx < HID * CPAD; idx += 256) {
        const int o = idx / CPAD, ch = idx % CPAD;
        w1s[o][ch] = (_Float16)(ch < CIN ? W1[o * CIN + ch] : 0.0f);
    }
    if (lane < KK) kid[w][lane] = kidx[((size_t)b * NN + n) * KK + lane];
    if (tid < HID) { ssum[tid] = 0.0f; ssq[tid] = 0.0f; }
    __syncthreads();

    // Stage 2: build the 16 x 160 fusion tile for this wave (channel-major fill).
    const float* pcb = pc1  + (size_t)b * 3  * NN;
    const float* ftb = feat1 + (size_t)b * DD * NN;
    for (int t = lane; t < CPAD * KK; t += 32) {
        const int ch  = t >> 4;
        const int row = t & 15;
        const int m   = kid[w][row];
        float v;
        if (ch < 3)         v = pcb[ch * NN + m] - pcb[ch * NN + n]; // xyz_diff
        else if (ch < 67)   v = ftb[(ch - 3)  * NN + m];             // neighbor feat
        else if (ch < CIN)  v = ftb[(ch - 67) * NN + n];             // center feat
        else                v = 0.0f;                                // pad
        fus[w][row][ch] = (_Float16)v;
    }
    __syncthreads();

    // Stage 3: 5 K-chunks x 2 column tiles of WMMA.
    v8f acc0 = {}, acc1 = {};
#pragma unroll
    for (int c = 0; c < 5; ++c) {
        v16h a, bf0, bf1;
#pragma unroll
        for (int e = 0; e < 16; ++e) {
            const int kk = c * 32 + ((e < 8) ? e : e + 8) + hi * 8; // ISA A/B striping
            a[e]   = fus[w][lo][kk];
            bf0[e] = w1s[lo][kk];
            bf1[e] = w1s[lo + 16][kk];
        }
        acc0 = __builtin_amdgcn_wmma_f32_16x16x32_f16(false, a, false, bf0,
                                                      (short)0, acc0, false, false);
        acc1 = __builtin_amdgcn_wmma_f32_16x16x32_f16(false, a, false, bf1,
                                                      (short)0, acc1, false, false);
    }

    // Stage 4: bias, store h, local stat accumulation.
    const float bias0 = b1[lo], bias1 = b1[lo + 16];
    float s0 = 0.f, q0 = 0.f, s1 = 0.f, q1 = 0.f;
    float* hp = hbuf + ((size_t)b * NN + n) * KK * HID;
#pragma unroll
    for (int r = 0; r < 8; ++r) {
        const int k = r + 8 * hi;       // C/D layout: row M = r + 8*(lane>=16)
        const float v0 = acc0[r] + bias0;
        const float v1 = acc1[r] + bias1;
        hp[(size_t)k * HID + lo]      = v0;
        hp[(size_t)k * HID + lo + 16] = v1;
        s0 += v0; q0 = fmaf(v0, v0, q0);
        s1 += v1; q1 = fmaf(v1, v1, q1);
    }
    atomicAdd(&ssum[lo], s0);      atomicAdd(&ssq[lo], q0);
    atomicAdd(&ssum[lo + 16], s1); atomicAdd(&ssq[lo + 16], q1);
    __syncthreads();
    if (tid < HID) {
        atomicAdd(&stats[tid], ssum[tid]);
        atomicAdd(&stats[HID + tid], ssq[tid]);
    }
}

// ---------------------------------------------------------------------------
// Kernel 3: fold BatchNorm stats into per-channel scale/shift.
// ---------------------------------------------------------------------------
__global__ void bn_fold_kernel(const float* __restrict__ stats,
                               const float* __restrict__ gamma,
                               const float* __restrict__ beta,
                               float* __restrict__ scbuf) {
    const int o = threadIdx.x;
    const float cnt  = (float)((size_t)BB * NN * KK);
    const float mean = stats[o] / cnt;
    const float var  = stats[HID + o] / cnt - mean * mean;
    const float sc   = gamma[o] * rsqrtf(var + EPSV);
    scbuf[o]        = sc;
    scbuf[HID + o]  = beta[o] - mean * sc;
}

// ---------------------------------------------------------------------------
// Kernel 4: BN + LeakyReLU + W2 + softmax over neighbors + flow re-weighting.
// One thread per (b,n).
// ---------------------------------------------------------------------------
__global__ void finalize_kernel(const float* __restrict__ hbuf,
                                const int*   __restrict__ kidx,
                                const float* __restrict__ flow,
                                const float* __restrict__ scbuf,
                                const float* __restrict__ W2,
                                const float* __restrict__ b2,
                                float* __restrict__ out) {
    __shared__ float sc[HID], sh[HID], w2s[HID];
    const int tid = threadIdx.x;
    if (tid < HID) { sc[tid] = scbuf[tid]; sh[tid] = scbuf[HID + tid]; w2s[tid] = W2[tid]; }
    __syncthreads();

    const int bn = blockIdx.x * 256 + tid;
    const int b  = bn >> 12;
    const int n  = bn & (NN - 1);
    const float b2v = b2[0];

    float logits[KK];
    const float* hp = hbuf + (size_t)bn * KK * HID;
#pragma unroll 4
    for (int k = 0; k < KK; ++k) {
        float acc = b2v;
        const float* hr = hp + (size_t)k * HID;
#pragma unroll
        for (int o = 0; o < HID; ++o) {
            float hv = fmaf(hr[o], sc[o], sh[o]);
            hv = (hv >= 0.0f) ? hv : SLOPE * hv;
            acc = fmaf(w2s[o], hv, acc);
        }
        logits[k] = acc;
    }
    float mx = logits[0];
#pragma unroll
    for (int k = 1; k < KK; ++k) mx = fmaxf(mx, logits[k]);
    float sum = 0.0f;
#pragma unroll
    for (int k = 0; k < KK; ++k) { logits[k] = __expf(logits[k] - mx); sum += logits[k]; }
    const float inv = 1.0f / sum;

    const float* flp = flow + (size_t)b * 3 * NN;
    const int*   kid = kidx + (size_t)bn * KK;
    float o0 = 0.f, o1 = 0.f, o2 = 0.f;
#pragma unroll 4
    for (int k = 0; k < KK; ++k) {
        const float wk = logits[k] * inv;
        const int m = kid[k];
        o0 = fmaf(wk, flp[m], o0);
        o1 = fmaf(wk, flp[NN + m], o1);
        o2 = fmaf(wk, flp[2 * NN + m], o2);
    }
    out[(size_t)bn * 3 + 0] = flp[n]          + o0;
    out[(size_t)bn * 3 + 1] = flp[NN + n]     + o1;
    out[(size_t)bn * 3 + 2] = flp[2 * NN + n] + o2;
}

// ---------------------------------------------------------------------------
extern "C" void kernel_launch(void* const* d_in, const int* in_sizes, int n_in,
                              void* d_out, int out_size, void* d_ws, size_t ws_size,
                              hipStream_t stream) {
    const float* pc1   = (const float*)d_in[0];
    const float* feat1 = (const float*)d_in[1];
    const float* flow  = (const float*)d_in[2];
    const float* W1    = (const float*)d_in[3];
    const float* b1    = (const float*)d_in[4];
    const float* gamma = (const float*)d_in[5];
    const float* beta  = (const float*)d_in[6];
    const float* W2    = (const float*)d_in[7];
    const float* b2    = (const float*)d_in[8];
    float* out = (float*)d_out;

    uint8_t* ws = (uint8_t*)d_ws;
    int*   kidx  = (int*)ws;                                  //   2 MiB
    float* hbuf  = (float*)(ws + (size_t)2097152);            //  64 MiB
    float* stats = (float*)(ws + (size_t)2097152 + 67108864); //  64 floats
    float* scbuf = stats + 64;                                //  64 floats

    // 1) KNN (8 batches x 16 blocks of 256 threads)
    knn_kernel<<<dim3(BB * 16), dim3(256), 0, stream>>>(pc1, kidx);

    // 2) zero stats accumulator
    hipMemsetAsync(stats, 0, 2 * HID * sizeof(float), stream);

    // 3) fusion + WMMA GEMM (8 waves/block, one wave per point)
    fusion_wmma_kernel<<<dim3((BB * NN) / 8), dim3(256), 0, stream>>>(
        pc1, feat1, kidx, W1, b1, hbuf, stats);

    // 4) fold BN stats
    bn_fold_kernel<<<dim3(1), dim3(HID), 0, stream>>>(stats, gamma, beta, scbuf);

    // 5) finalize
    finalize_kernel<<<dim3((BB * NN) / 256), dim3(256), 0, stream>>>(
        hbuf, kidx, flow, scbuf, W2, b2, out);
}
